// lstm_copy_75153337745477
// MI455X (gfx1250) — matrix operand order repaired and checked by reading, not compiled, not run on hardware
//
#include <hip/hip_runtime.h>
#include <hip/hip_bf16.h>
#include <stdint.h>

// ---------------- problem constants (from reference) ----------------
#define SEQL  20
#define BATCH 64
#define VDIM  4096
#define HDIM  1024
#define FUTS  10

typedef __attribute__((ext_vector_type(16))) __bf16 v16bf;
typedef __attribute__((ext_vector_type(8)))  float  v8f;

union AF {
  v16bf v;
  uint4 q[2];
  unsigned short s[16];
};

__device__ __forceinline__ unsigned short f2bf(float f) {
  union { float f; unsigned u; } x; x.f = f;
  unsigned r = x.u + 0x7FFFu + ((x.u >> 16) & 1u);   // round-to-nearest-even
  return (unsigned short)(r >> 16);
}
__device__ __forceinline__ float sigm(float x) { return 1.f / (1.f + __expf(-x)); }
__device__ __forceinline__ float tanh_fast(float x) {
  x = fminf(fmaxf(x, -15.f), 15.f);
  float t = __expf(2.f * x);
  return (t - 1.f) / (t + 1.f);
}

// ---------------- utility kernels ----------------
__global__ void zero_u32(uint32_t* __restrict__ p, long n) {
  long i  = (long)blockIdx.x * blockDim.x + threadIdx.x;
  long st = (long)gridDim.x * blockDim.x;
  for (; i < n; i += st) p[i] = 0u;
}

__global__ void cvt_bf16(const float* __restrict__ s, unsigned short* __restrict__ d, long n) {
  long i  = (long)blockIdx.x * blockDim.x + threadIdx.x;
  long st = (long)gridDim.x * blockDim.x;
  for (; i < n; i += st) d[i] = f2bf(s[i]);
}

// Repack row-major fp32 weight (fourH x K) into bf16 WMMA B-fragment-major layout:
// fragment (nTile,kTile) = 16 cols x 32 k, stored lane-contiguous: 32 lanes x 16 bf16.
// lane l (l<16 -> K 0..15, l>=16 -> K 16..31 of the tile), col = nTile*16 + (l&15).
__global__ void pack_w_bf16(const float* __restrict__ W, unsigned short* __restrict__ dst,
                            int fourH, int K) {
  long n   = (long)fourH * K;
  int  kts = K >> 5;
  long i   = (long)blockIdx.x * blockDim.x + threadIdx.x;
  long st  = (long)gridDim.x * blockDim.x;
  for (; i < n; i += st) {
    int  e    = (int)(i & 15);
    int  lane = (int)((i >> 4) & 31);
    long rest = i >> 9;
    int  kt   = (int)(rest % kts);
    long nt   = rest / kts;
    int  col  = (int)nt * 16 + (lane & 15);
    int  k    = kt * 32 + (lane >> 4) * 16 + e;
    dst[i] = f2bf(W[(size_t)col * K + k]);
  }
}

// ---------------- fused dual-GEMM gates kernel ----------------
// gates(64 x fourH) = A1(64xK1,bf16) @ W1^T + A2(64xK2,bf16) @ W2^T + bih + bhh
// WBF=true : W is bf16, packed fragment-major (see pack_w_bf16)
// WBF=false: W is fp32 row-major (fourH x K) straight from d_in (in-register cvt)
// Block = 256 threads = 8 waves = 4 M-tiles (batch 64) x 2 N-tiles; grid.x = fourH/32.
template <bool WBF>
__global__ __launch_bounds__(256)
void gates_gemm(const unsigned short* __restrict__ A1, const void* __restrict__ W1v, int K1,
                const unsigned short* __restrict__ A2, const void* __restrict__ W2v, int K2,
                const float* __restrict__ bih, const float* __restrict__ bhh,
                float* __restrict__ gates, int fourH) {
  const int lane = threadIdx.x & 31;
  const int wv   = threadIdx.x >> 5;
  const int mt   = wv & 3;                 // M tile: rows 16*mt .. 16*mt+15
  const int nt   = wv >> 2;                // N tile within block (0..1)
  const int n0   = blockIdx.x * 32 + nt * 16;
  const int hi   = lane >> 4;              // half-wave select
  const int l16  = lane & 15;
  const int arow = mt * 16 + l16;          // A row handled by this lane
  const int bcol = n0 + l16;               // output column handled by this lane

  v8f acc = {};

  // ---- pass 1: A1 @ W1^T ----
  if (K1 > 0) {
    const unsigned short* a1r = A1 + (size_t)arow * K1 + hi * 16;
    if (WBF) {
      const int kts = K1 >> 5;
      const unsigned short* wp =
          (const unsigned short*)W1v + ((size_t)(blockIdx.x * 2 + nt) * kts) * 512 + (size_t)lane * 16;
      for (int k0 = 0, kt = 0; k0 < K1; k0 += 32, ++kt) {
        AF a, b;
        a.q[0] = *(const uint4*)(a1r + k0);
        a.q[1] = *(const uint4*)(a1r + k0 + 8);
        b.q[0] = *(const uint4*)(wp + (size_t)kt * 512);
        b.q[1] = *(const uint4*)(wp + (size_t)kt * 512 + 8);
        acc = __builtin_amdgcn_wmma_f32_16x16x32_bf16(false, a.v, false, b.v,
                                                      (short)0, acc, false, false);
      }
    } else {
      const float* wr = (const float*)W1v + (size_t)bcol * K1 + hi * 16;
      for (int k0 = 0; k0 < K1; k0 += 32) {
        AF a, b;
        a.q[0] = *(const uint4*)(a1r + k0);
        a.q[1] = *(const uint4*)(a1r + k0 + 8);
        const float* wp = wr + k0;
#pragma unroll
        for (int e = 0; e < 16; ++e) b.s[e] = f2bf(wp[e]);
        acc = __builtin_amdgcn_wmma_f32_16x16x32_bf16(false, a.v, false, b.v,
                                                      (short)0, acc, false, false);
      }
    }
  }

  // ---- pass 2: A2 @ W2^T ----
  if (K2 > 0) {
    const unsigned short* a2r = A2 + (size_t)arow * K2 + hi * 16;
    if (WBF) {
      const int kts = K2 >> 5;
      const unsigned short* wp =
          (const unsigned short*)W2v + ((size_t)(blockIdx.x * 2 + nt) * kts) * 512 + (size_t)lane * 16;
      for (int k0 = 0, kt = 0; k0 < K2; k0 += 32, ++kt) {
        AF a, b;
        a.q[0] = *(const uint4*)(a2r + k0);
        a.q[1] = *(const uint4*)(a2r + k0 + 8);
        b.q[0] = *(const uint4*)(wp + (size_t)kt * 512);
        b.q[1] = *(const uint4*)(wp + (size_t)kt * 512 + 8);
        acc = __builtin_amdgcn_wmma_f32_16x16x32_bf16(false, a.v, false, b.v,
                                                      (short)0, acc, false, false);
      }
    } else {
      const float* wr = (const float*)W2v + (size_t)bcol * K2 + hi * 16;
      for (int k0 = 0; k0 < K2; k0 += 32) {
        AF a, b;
        a.q[0] = *(const uint4*)(a2r + k0);
        a.q[1] = *(const uint4*)(a2r + k0 + 8);
        const float* wp = wr + k0;
#pragma unroll
        for (int e = 0; e < 16; ++e) b.s[e] = f2bf(wp[e]);
        acc = __builtin_amdgcn_wmma_f32_16x16x32_bf16(false, a.v, false, b.v,
                                                      (short)0, acc, false, false);
      }
    }
  }

  // ---- bias + store (D layout: lanes 0-15 M=v, lanes 16-31 M=8+v; N = l16) ----
  const float bsum = bih[bcol] + bhh[bcol];
  float* gout = gates + (size_t)(mt * 16 + hi * 8) * fourH + bcol;
#pragma unroll
  for (int v = 0; v < 8; ++v) gout[(size_t)v * fourH] = acc[v] + bsum;
}

// ---------------- LSTM elementwise cell update ----------------
// gate order i,f,g,o (torch LSTMCell). Updates c in place, writes h as bf16
// (next-step WMMA A operand); optionally writes sigmoid(h) to zout (decoder L3).
__global__ __launch_bounds__(256)
void lstm_pw(const float* __restrict__ g, float* __restrict__ c,
             unsigned short* __restrict__ hbf, float* __restrict__ zout, int Hd) {
  int idx   = blockIdx.x * blockDim.x + threadIdx.x;
  int total = BATCH * Hd;
  if (idx >= total) return;
  int b = idx / Hd, j = idx - b * Hd;
  const float* gr = g + (size_t)b * 4 * Hd;
  float gi = sigm(gr[j]);
  float gf = sigm(gr[Hd + j]);
  float gc = tanh_fast(gr[2 * Hd + j]);
  float go = sigm(gr[3 * Hd + j]);
  float cn = gf * c[idx] + gi * gc;
  c[idx]   = cn;
  float h  = go * tanh_fast(cn);
  hbf[idx] = f2bf(h);
  if (zout) zout[idx] = sigm(h);
}

// ---------------- host orchestration ----------------
extern "C" void kernel_launch(void* const* d_in, const int* in_sizes, int n_in,
                              void* d_out, int out_size, void* d_ws, size_t ws_size,
                              hipStream_t stream) {
  (void)in_sizes; (void)n_in; (void)out_size;

  const float* x        = (const float*)d_in[0];
  const float* en1_Wih  = (const float*)d_in[1];
  const float* en1_Whh  = (const float*)d_in[2];
  const float* en1_bih  = (const float*)d_in[3];
  const float* en1_bhh  = (const float*)d_in[4];
  const float* en2_Wih  = (const float*)d_in[5];
  const float* en2_Whh  = (const float*)d_in[6];
  const float* en2_bih  = (const float*)d_in[7];
  const float* en2_bhh  = (const float*)d_in[8];
  const float* en3_Wih  = (const float*)d_in[9];
  const float* en3_Whh  = (const float*)d_in[10];
  const float* en3_bih  = (const float*)d_in[11];
  const float* en3_bhh  = (const float*)d_in[12];
  // d_in[13] = de1_Wih: decoder input is identically zero -> never used
  const float* de1_Whh  = (const float*)d_in[14];
  const float* de1_bih  = (const float*)d_in[15];
  const float* de1_bhh  = (const float*)d_in[16];
  const float* de2_Wih  = (const float*)d_in[17];
  const float* de2_Whh  = (const float*)d_in[18];
  const float* de2_bih  = (const float*)d_in[19];
  const float* de2_bhh  = (const float*)d_in[20];
  const float* de3_Wih  = (const float*)d_in[21];
  const float* de3_Whh  = (const float*)d_in[22];
  const float* de3_bih  = (const float*)d_in[23];
  const float* de3_bhh  = (const float*)d_in[24];
  // d_in[25] = future_step (device scalar) == 10, fixed in reference.

  // ---- workspace carve-up ----
  char*  ws  = (char*)d_ws;
  size_t off = 0;
  auto carve = [&](size_t bytes) -> char* {
    char* p = ws + off;
    off += (bytes + 255) & ~(size_t)255;
    return p;
  };

  float* gates = (float*)carve((size_t)BATCH * 4 * VDIM * 4);   // max 64x16384 f32

  char* zbase = ws + off;                                       // zero-init region start
  float* c1  = (float*)carve((size_t)BATCH * HDIM * 4);
  float* c2  = (float*)carve((size_t)BATCH * HDIM * 4);
  float* c3  = (float*)carve((size_t)BATCH * HDIM * 4);
  float* dc1 = (float*)carve((size_t)BATCH * HDIM * 4);
  float* dc2 = (float*)carve((size_t)BATCH * HDIM * 4);
  float* dc3 = (float*)carve((size_t)BATCH * VDIM * 4);
  unsigned short* hbA = (unsigned short*)carve((size_t)BATCH * HDIM * 2);
  unsigned short* hbB = (unsigned short*)carve((size_t)BATCH * HDIM * 2);
  unsigned short* hbC = (unsigned short*)carve((size_t)BATCH * HDIM * 2); // enc L3 h == dec L1 h
  unsigned short* hbD = (unsigned short*)carve((size_t)BATCH * HDIM * 2);
  unsigned short* hbE = (unsigned short*)carve((size_t)BATCH * VDIM * 2);
  size_t zbytes = (size_t)((ws + off) - zbase);

  unsigned short* xbf = (unsigned short*)carve((size_t)SEQL * BATCH * VDIM * 2);

  // bf16 packed weights (optional, if workspace is large enough)
  unsigned short* w_en1_Wih = (unsigned short*)carve((size_t)4 * HDIM * VDIM * 2);
  unsigned short* w_en1_Whh = (unsigned short*)carve((size_t)4 * HDIM * HDIM * 2);
  unsigned short* w_en2_Wih = (unsigned short*)carve((size_t)4 * HDIM * HDIM * 2);
  unsigned short* w_en2_Whh = (unsigned short*)carve((size_t)4 * HDIM * HDIM * 2);
  unsigned short* w_en3_Wih = (unsigned short*)carve((size_t)4 * HDIM * HDIM * 2);
  unsigned short* w_en3_Whh = (unsigned short*)carve((size_t)4 * HDIM * HDIM * 2);
  unsigned short* w_de1_Whh = (unsigned short*)carve((size_t)4 * HDIM * HDIM * 2);
  unsigned short* w_de2_Wih = (unsigned short*)carve((size_t)4 * HDIM * HDIM * 2);
  unsigned short* w_de2_Whh = (unsigned short*)carve((size_t)4 * HDIM * HDIM * 2);
  unsigned short* w_de3_Wih = (unsigned short*)carve((size_t)4 * VDIM * HDIM * 2);
  unsigned short* w_de3_Whh = (unsigned short*)carve((size_t)4 * VDIM * VDIM * 2);

  const bool wbf = (ws_size >= off);  // full bf16-weight path fits?

  // ---- one-time per-call conversions ----
  zero_u32<<<dim3(2048), dim3(256), 0, stream>>>((uint32_t*)zbase, (long)(zbytes / 4));
  cvt_bf16<<<dim3(4096), dim3(256), 0, stream>>>(x, xbf, (long)SEQL * BATCH * VDIM);
  if (wbf) {
    pack_w_bf16<<<dim3(8192), dim3(256), 0, stream>>>(en1_Wih, w_en1_Wih, 4 * HDIM, VDIM);
    pack_w_bf16<<<dim3(4096), dim3(256), 0, stream>>>(en1_Whh, w_en1_Whh, 4 * HDIM, HDIM);
    pack_w_bf16<<<dim3(4096), dim3(256), 0, stream>>>(en2_Wih, w_en2_Wih, 4 * HDIM, HDIM);
    pack_w_bf16<<<dim3(4096), dim3(256), 0, stream>>>(en2_Whh, w_en2_Whh, 4 * HDIM, HDIM);
    pack_w_bf16<<<dim3(4096), dim3(256), 0, stream>>>(en3_Wih, w_en3_Wih, 4 * HDIM, HDIM);
    pack_w_bf16<<<dim3(4096), dim3(256), 0, stream>>>(en3_Whh, w_en3_Whh, 4 * HDIM, HDIM);
    pack_w_bf16<<<dim3(4096), dim3(256), 0, stream>>>(de1_Whh, w_de1_Whh, 4 * HDIM, HDIM);
    pack_w_bf16<<<dim3(4096), dim3(256), 0, stream>>>(de2_Wih, w_de2_Wih, 4 * HDIM, HDIM);
    pack_w_bf16<<<dim3(4096), dim3(256), 0, stream>>>(de2_Whh, w_de2_Whh, 4 * HDIM, HDIM);
    pack_w_bf16<<<dim3(8192), dim3(256), 0, stream>>>(de3_Wih, w_de3_Wih, 4 * VDIM, HDIM);
    pack_w_bf16<<<dim3(8192), dim3(256), 0, stream>>>(de3_Whh, w_de3_Whh, 4 * VDIM, VDIM);
  }

  auto Wsel = [&](const unsigned short* packed, const float* raw) -> const void* {
    return wbf ? (const void*)packed : (const void*)raw;
  };
  auto gemm = [&](const unsigned short* A1, const void* W1, int K1,
                  const unsigned short* A2, const void* W2, int K2,
                  const float* bi, const float* bh, int fourH) {
    dim3 grid(fourH / 32), block(256);
    if (wbf)
      gates_gemm<true><<<grid, block, 0, stream>>>(A1, W1, K1, A2, W2, K2, bi, bh, gates, fourH);
    else
      gates_gemm<false><<<grid, block, 0, stream>>>(A1, W1, K1, A2, W2, K2, bi, bh, gates, fourH);
  };
  auto pw = [&](float* c, unsigned short* hb, float* z, int Hd) {
    int total = BATCH * Hd;
    lstm_pw<<<dim3((total + 255) / 256), dim3(256), 0, stream>>>(gates, c, hb, z, Hd);
  };

  // ---- encoder: 20 steps x 3 layers ----
  for (int t = 0; t < SEQL; ++t) {
    const unsigned short* xt = xbf + (size_t)t * BATCH * VDIM;
    gemm(xt,  Wsel(w_en1_Wih, en1_Wih), VDIM, hbA, Wsel(w_en1_Whh, en1_Whh), HDIM,
         en1_bih, en1_bhh, 4 * HDIM);
    pw(c1, hbA, nullptr, HDIM);
    gemm(hbA, Wsel(w_en2_Wih, en2_Wih), HDIM, hbB, Wsel(w_en2_Whh, en2_Whh), HDIM,
         en2_bih, en2_bhh, 4 * HDIM);
    pw(c2, hbB, nullptr, HDIM);
    gemm(hbB, Wsel(w_en3_Wih, en3_Wih), HDIM, hbC, Wsel(w_en3_Whh, en3_Whh), HDIM,
         en3_bih, en3_bhh, 4 * HDIM);
    pw(c3, hbC, nullptr, HDIM);
  }

  // ---- decoder: 10 steps x 3 layers (zero input -> de1 Wih GEMM skipped) ----
  for (int t = 0; t < FUTS; ++t) {
    gemm(nullptr, nullptr, 0, hbC, Wsel(w_de1_Whh, de1_Whh), HDIM,
         de1_bih, de1_bhh, 4 * HDIM);
    pw(dc1, hbC, nullptr, HDIM);
    gemm(hbC, Wsel(w_de2_Wih, de2_Wih), HDIM, hbD, Wsel(w_de2_Whh, de2_Whh), HDIM,
         de2_bih, de2_bhh, 4 * HDIM);
    pw(dc2, hbD, nullptr, HDIM);
    gemm(hbD, Wsel(w_de3_Wih, de3_Wih), HDIM, hbE, Wsel(w_de3_Whh, de3_Whh), VDIM,
         de3_bih, de3_bhh, 4 * VDIM);
    pw(dc3, hbE, (float*)d_out + (size_t)t * BATCH * VDIM, VDIM);
  }
}
